// BottomEncoder_47184510714071
// MI455X (gfx1250) — compile-verified, matrix operand order checked by reading, not run on hardware
//
#include <hip/hip_runtime.h>

// ---------------- problem constants (match reference) ----------------
#define NATOMS 50000
#define NEDGES 800000
#define HID    128
#define HEADS  4
#define FHEAD  32

typedef __attribute__((ext_vector_type(16))) _Float16 v16h;
typedef __attribute__((ext_vector_type(8)))  float    v8f;

// ---------------- operand packing into WMMA lane layout ----------------
// ISA 7.12.2, 16-bit A 16x32 (one 32-wide k-step):
//   lane = (row&15) | (bit3(kk)<<4); pair base = kk&0b10110;
//   j = base<8 ? base/2 : (base-16)/2+4; slot = 2j + (kk&1).
// Packed layout: [tile][kstep 0..3][lane 0..31][16 halves]  (32B per lane frag)
__device__ __forceinline__ int frag_slot(int kk) {
  int base = kk & 22;  // keep bits 4,2,1
  int j = (base >= 16) ? (((base - 16) >> 1) + 4) : (base >> 1);
  return 2 * j + (kk & 1);
}

// A: in is [M,128] row-major f32 (tile index from row)
__global__ void pack_a(const float* __restrict__ in, _Float16* __restrict__ out, int n) {
  int i = blockIdx.x * blockDim.x + threadIdx.x;
  if (i >= n) return;
  int m = i >> 7, k = i & 127;
  int kstep = k >> 5, kk = k & 31;
  int lane = (m & 15) | (((kk >> 3) & 1) << 4);
  size_t d = ((size_t)((m >> 4) * 4 + kstep) * 32 + lane) * 16 + frag_slot(kk);
  out[d] = (_Float16)in[i];
}

// B: in is [128 (K), 128 (N)] row-major f32 (tile index from column)
__global__ void pack_b(const float* __restrict__ in, _Float16* __restrict__ out, int n) {
  int i = blockIdx.x * blockDim.x + threadIdx.x;
  if (i >= n) return;
  int k = i >> 7, ncol = i & 127;
  int kstep = k >> 5, kk = k & 31;
  int lane = (ncol & 15) | (((kk >> 3) & 1) << 4);
  size_t d = ((size_t)((ncol >> 4) * 4 + kstep) * 32 + lane) * 16 + frag_slot(kk);
  out[d] = (_Float16)in[i];
}

// ---------------- small utility kernels ----------------
__global__ void fill_f32(float* __restrict__ p, float v, int n) {
  int i = blockIdx.x * blockDim.x + threadIdx.x;
  if (i < n) p[i] = v;
}

__global__ void copy_f32(const float* __restrict__ in, float* __restrict__ out, int n) {
  int i = blockIdx.x * blockDim.x + threadIdx.x;
  if (i < n) out[i] = in[i];
}

// ---------------- WMMA GEMM on pre-packed operands ----------------
// One wave computes one 16x16 tile of C = A[M,128] x B[128,128].
// Inner loop: 2x global_load_b128 (A frag) + 2x (B frag) + 1 wmma per k-step.
__global__ __launch_bounds__(32) void wmma_gemm128(
    const v16h* __restrict__ Ap, const v16h* __restrict__ Bp,
    float* __restrict__ C, int ldc, int colOff,
    const float* __restrict__ bias, int relu_act) {
  const int lane = threadIdx.x;
  const int tm = blockIdx.x;
  const int tn = blockIdx.y;
  const v16h* __restrict__ a0 = Ap + ((size_t)tm * 4) * 32 + lane;
  const v16h* __restrict__ b0 = Bp + ((size_t)tn * 4) * 32 + lane;

  v8f c = {0.f, 0.f, 0.f, 0.f, 0.f, 0.f, 0.f, 0.f};
#pragma unroll
  for (int ks = 0; ks < 4; ++ks) {
    v16h a = a0[ks * 32];
    v16h b = b0[ks * 32];
    // v_wmma_f32_16x16x32_f16: (neg_a, A, neg_b, B, c_mod, C, reuse_a, reuse_b)
    c = __builtin_amdgcn_wmma_f32_16x16x32_f16(false, a, false, b, (short)0, c,
                                               false, false);
  }

  // C/D layout: VGPR r -> row r (lanes 0-15) / row r+8 (lanes 16-31), col = lane&15
  const int row0 = tm * 16 + ((lane >> 4) << 3);
  const int col  = tn * 16 + (lane & 15) + colOff;
  const float bv = bias ? bias[tn * 16 + (lane & 15)] : 0.0f;
#pragma unroll
  for (int r = 0; r < 8; ++r) {
    float v = c[r] + bv;
    if (relu_act) v = fmaxf(v, 0.0f);
    C[(size_t)(row0 + r) * ldc + col] = v;
  }
}

// ---------------- attention logits: el/er [N,HEADS] ----------------
__global__ void attn_logits(const float* __restrict__ z,
                            const float* __restrict__ al, const float* __restrict__ ar,
                            float* __restrict__ el, float* __restrict__ er, int nh) {
  int i = blockIdx.x * blockDim.x + threadIdx.x;
  if (i >= nh) return;
  int n = i >> 2, h = i & 3;
  const float* zp  = z  + (size_t)n * HID + h * FHEAD;
  const float* alp = al + h * FHEAD;
  const float* arp = ar + h * FHEAD;
  float sl = 0.f, sr = 0.f;
#pragma unroll
  for (int f = 0; f < FHEAD; ++f) {
    float zv = zp[f];
    sl += zv * alp[f];
    sr += zv * arp[f];
  }
  el[i] = sl;
  er[i] = sr;
}

// ---------------- float atomic max via CAS ----------------
__device__ __forceinline__ void atomicMaxF(float* addr, float val) {
  unsigned* ua = (unsigned*)addr;
  unsigned old = *ua;
  while (__uint_as_float(old) < val) {
    unsigned assumed = old;
    old = atomicCAS(ua, assumed, __float_as_uint(val));
    if (old == assumed) break;
  }
}

__device__ __forceinline__ float leaky02(float x) { return (x > 0.f) ? x : 0.2f * x; }

// ---------------- edge softmax passes ----------------
__global__ void edge_attn_max(const int* __restrict__ src, const int* __restrict__ dst,
                              const float* __restrict__ el, const float* __restrict__ er,
                              float* __restrict__ mmax, int e4) {
  int i = blockIdx.x * blockDim.x + threadIdx.x;
  if (i >= e4) return;
  int e = i >> 2, h = i & 3;
  int s = src[e], d = dst[e];
  float x = leaky02(el[s * HEADS + h] + er[d * HEADS + h]);
  atomicMaxF(&mmax[d * HEADS + h], x);
}

__global__ void edge_exp_sum(const int* __restrict__ src, const int* __restrict__ dst,
                             const float* __restrict__ el, const float* __restrict__ er,
                             const float* __restrict__ mmax,
                             float* __restrict__ aexp, float* __restrict__ denom, int e4) {
  int i = blockIdx.x * blockDim.x + threadIdx.x;
  if (i >= e4) return;
  int e = i >> 2, h = i & 3;
  int s = src[e], d = dst[e];
  float x = leaky02(el[s * HEADS + h] + er[d * HEADS + h]);
  float a = __expf(x - mmax[d * HEADS + h]);
  aexp[i] = a;
  atomicAdd(&denom[d * HEADS + h], a);
}

// msg scatter: acc[dst, c] += alpha(e,h) * z[src, c]   (c = h*32+f)
__global__ void edge_scatter_msg(const int* __restrict__ src, const int* __restrict__ dst,
                                 const float* __restrict__ z,
                                 const float* __restrict__ aexp, const float* __restrict__ denom,
                                 float* __restrict__ acc, int ec) {
  int i = blockIdx.x * blockDim.x + threadIdx.x;
  if (i >= ec) return;
  int e = i >> 7, c = i & 127, h = c >> 5;
  int s = src[e], d = dst[e];
  float alpha = aexp[(size_t)e * HEADS + h] / denom[d * HEADS + h];
  atomicAdd(&acc[(size_t)d * HID + c], alpha * z[(size_t)s * HID + c]);
}

// acc = relu(acc + bias), in place
__global__ void gat_finalize(float* __restrict__ acc, const float* __restrict__ bias, int n) {
  int i = blockIdx.x * blockDim.x + threadIdx.x;
  if (i >= n) return;
  acc[i] = fmaxf(acc[i] + bias[i & 127], 0.0f);
}

// GIN sum aggregation: agg[dst, c] += h[src, c]
__global__ void edge_sum(const int* __restrict__ src, const int* __restrict__ dst,
                         const float* __restrict__ h, float* __restrict__ agg, int ec) {
  int i = blockIdx.x * blockDim.x + threadIdx.x;
  if (i >= ec) return;
  int e = i >> 7, c = i & 127;
  atomicAdd(&agg[(size_t)dst[e] * HID + c], h[(size_t)src[e] * HID + c]);
}

// ---------------- workspace layout (bytes, 256-aligned) ----------------
static constexpr size_t al256(size_t x) { return (x + 255) & ~(size_t)255; }
static constexpr size_t OFF_APACK  = 0;
static constexpr size_t OFF_BPACK  = OFF_APACK + al256((size_t)NATOMS * HID * 2);
static constexpr size_t OFF_Z      = OFF_BPACK + al256((size_t)HID * HID * 2);
static constexpr size_t OFF_EL     = OFF_Z     + al256((size_t)NATOMS * HID * 4);
static constexpr size_t OFF_ER     = OFF_EL    + al256((size_t)NATOMS * HEADS * 4);
static constexpr size_t OFF_MMAX   = OFF_ER    + al256((size_t)NATOMS * HEADS * 4);
static constexpr size_t OFF_DENOM  = OFF_MMAX  + al256((size_t)NATOMS * HEADS * 4);
static constexpr size_t OFF_AEXP   = OFF_DENOM + al256((size_t)NATOMS * HEADS * 4);
static constexpr size_t OFF_ACC    = OFF_AEXP  + al256((size_t)NEDGES * HEADS * 4);
static constexpr size_t OFF_AGG    = OFF_ACC   + al256((size_t)NATOMS * HID * 4);
static constexpr size_t OFF_AGGP   = OFF_AGG   + al256((size_t)NATOMS * HID * 4);

extern "C" void kernel_launch(void* const* d_in, const int* in_sizes, int n_in,
                              void* d_out, int out_size, void* d_ws, size_t ws_size,
                              hipStream_t stream) {
  (void)in_sizes; (void)n_in; (void)out_size; (void)ws_size;

  const float* feats    = (const float*)d_in[0];
  const int*   src[2]   = {(const int*)d_in[1], (const int*)d_in[3]};
  const int*   dst[2]   = {(const int*)d_in[2], (const int*)d_in[4]};
  const float* Wgat[2]  = {(const float*)d_in[5],  (const float*)d_in[11]};
  const float* al[2]    = {(const float*)d_in[6],  (const float*)d_in[12]};
  const float* ar[2]    = {(const float*)d_in[7],  (const float*)d_in[13]};
  const float* bgat[2]  = {(const float*)d_in[8],  (const float*)d_in[14]};
  const float* Wgin[2]  = {(const float*)d_in[9],  (const float*)d_in[15]};
  const float* bgin[2]  = {(const float*)d_in[10], (const float*)d_in[16]};
  float* out = (float*)d_out;

  char* ws = (char*)d_ws;
  _Float16* apack = (_Float16*)(ws + OFF_APACK);
  _Float16* bpack = (_Float16*)(ws + OFF_BPACK);
  float*    z     = (float*)   (ws + OFF_Z);
  float*    el    = (float*)   (ws + OFF_EL);
  float*    er    = (float*)   (ws + OFF_ER);
  float*    mmax  = (float*)   (ws + OFF_MMAX);
  float*    denom = (float*)   (ws + OFF_DENOM);
  float*    aexp  = (float*)   (ws + OFF_AEXP);
  float*    acc   = (float*)   (ws + OFF_ACC);
  float*    agg   = (float*)   (ws + OFF_AGG);
  _Float16* aggp  = (_Float16*)(ws + OFF_AGGP);

  const int NC = NATOMS * HID;       // 6,400,000
  const int NH = NATOMS * HEADS;     //   200,000
  const int E4 = NEDGES * HEADS;     // 3,200,000
  const int EC = NEDGES * HID;       // 102,400,000 (< 2^31)
  const int WC = HID * HID;          // 16,384
  auto nb = [](int n, int b) { return (n + b - 1) / b; };

  // one-time: features -> f16, pre-packed into WMMA A-operand lane layout
  pack_a<<<nb(NC, 256), 256, 0, stream>>>(feats, apack, NC);

  for (int t = 0; t < 2; ++t) {
    // ---- GAT ----
    // z = feats @ W_gat  (WMMA f16->f32, both operands pre-packed)
    pack_b<<<nb(WC, 256), 256, 0, stream>>>(Wgat[t], bpack, WC);
    wmma_gemm128<<<dim3(NATOMS / 16, HID / 16), 32, 0, stream>>>(
        (const v16h*)apack, (const v16h*)bpack, z, HID, 0, nullptr, 0);

    attn_logits<<<nb(NH, 256), 256, 0, stream>>>(z, al[t], ar[t], el, er, NH);

    fill_f32<<<nb(NH, 256), 256, 0, stream>>>(mmax, -3.0e38f, NH);
    fill_f32<<<nb(NH, 256), 256, 0, stream>>>(denom, 0.0f, NH);
    fill_f32<<<nb(NC, 256), 256, 0, stream>>>(acc, 0.0f, NC);

    edge_attn_max<<<nb(E4, 256), 256, 0, stream>>>(src[t], dst[t], el, er, mmax, E4);
    edge_exp_sum<<<nb(E4, 256), 256, 0, stream>>>(src[t], dst[t], el, er, mmax,
                                                  aexp, denom, E4);
    edge_scatter_msg<<<nb(EC, 256), 256, 0, stream>>>(src[t], dst[t], z, aexp,
                                                      denom, acc, EC);
    gat_finalize<<<nb(NC, 256), 256, 0, stream>>>(acc, bgat[t], NC);

    // ---- GIN ----
    // agg = (1+eps)*h + sum_neigh h   (eps = 0)
    copy_f32<<<nb(NC, 256), 256, 0, stream>>>(acc, agg, NC);
    edge_sum<<<nb(EC, 256), 256, 0, stream>>>(src[t], dst[t], acc, agg, EC);
    pack_a<<<nb(NC, 256), 256, 0, stream>>>(agg, aggp, NC);

    // out[:, t*128:(t+1)*128] = relu(agg @ W_gin + b_gin)
    // (relu . leaky_relu = relu; outer relu idempotent)
    pack_b<<<nb(WC, 256), 256, 0, stream>>>(Wgin[t], bpack, WC);
    wmma_gemm128<<<dim3(NATOMS / 16, HID / 16), 32, 0, stream>>>(
        (const v16h*)aggp, (const v16h*)bpack, out, 2 * HID, t * HID, bgin[t], 1);
  }
}
